// GraphAttention_18734647345746
// MI455X (gfx1250) — matrix operand order repaired and checked by reading, not compiled, not run on hardware
//
#include <hip/hip_runtime.h>
#include <hip/hip_bf16.h>

typedef __attribute__((ext_vector_type(16))) _Float16 v16h;
typedef __attribute__((ext_vector_type(8)))  _Float16 v8h;
typedef __attribute__((ext_vector_type(8)))  float    v8f;

#define NNODES 6144
#define NHEADS 3

// ---------------------------------------------------------------------------
// WMMA helper: D = A(16x32 f16) x B(32x16 f16) + C(16x16 f32)
// ---------------------------------------------------------------------------
__device__ __forceinline__ v8f wmma_f16(v16h a, v16h b, v8f c) {
    return __builtin_amdgcn_wmma_f32_16x16x32_f16(
        /*neg_a=*/false, a, /*neg_b=*/false, b,
        /*c_mod=*/(short)0, c, /*reuse_a=*/false, /*reuse_b=*/false);
}

__device__ __forceinline__ v16h cat8(v8h lo, v8h hi) {
    return __builtin_shufflevector(lo, hi, 0,1,2,3,4,5,6,7,8,9,10,11,12,13,14,15);
}

// ---------------------------------------------------------------------------
// Prep: f32 -> f16 elementwise convert (feature matrix).
// ---------------------------------------------------------------------------
__global__ void cvt_f16(const float* __restrict__ x, _Float16* __restrict__ y, int count)
{
    int i = blockIdx.x * blockDim.x + threadIdx.x;
    if (i < count) y[i] = (_Float16)x[i];
}

// ---------------------------------------------------------------------------
// Prep: W[h,k,col(AD)] -> WT_f16[h,col(FOUT,padded),k]  (zero-pad cols >= AD)
// ---------------------------------------------------------------------------
template<int FIN, int FOUT, int AD>
__global__ void wt_prep(const float* __restrict__ W, _Float16* __restrict__ WT)
{
    int idx = blockIdx.x * blockDim.x + threadIdx.x;   // over H*FOUT*FIN
    if (idx >= NHEADS * FOUT * FIN) return;
    const int k   = idx % FIN;
    const int col = (idx / FIN) % FOUT;
    const int h   = idx / (FIN * FOUT);
    float v = (col < AD) ? W[((size_t)h * FIN + k) * AD + col] : 0.0f;
    WT[idx] = (_Float16)v;
}

// ---------------------------------------------------------------------------
// Kernel 1: Wh[h,n,o] = sum_f x[n,f] * W[h,f,o]   (one wave per 16-row tile).
// A lane layout (16-bit A 16x32): lane l -> row l&15; K = kb+j (j=0..7) and
// kb+8+j, kb = 16*(l>=16) -> two 16B v8h loads from x16.
// B lane layout (16-bit B 32x16): lane l -> col l&15; K = kB..kB+15,
// kB = 16*(l>=16) -> one 32B v16h load from WT (transposed f16).
// C layout: vgpr r, lane l -> row r + 8*(l>=16), col l&15.
// Writes Wh (f32) and WhT_f16[h,o,n] for the aggregation B operands.
// ---------------------------------------------------------------------------
template<int FIN, int FOUT>
__global__ __launch_bounds__(32) void wh_gemm(
    const _Float16* __restrict__ x16,  // [N, FIN]  f16
    const _Float16* __restrict__ WT,   // [H, FOUT, FIN] f16 (padded)
    float*          __restrict__ Wh,   // [H, N, FOUT]
    _Float16*       __restrict__ WhT,  // [H, FOUT, N]
    int n)
{
    const int tile = blockIdx.x;
    const int h    = blockIdx.y;
    const int lane = threadIdx.x;
    const int rlo  = lane & 15;
    const int hi   = lane >> 4;
    const int rowA = tile * 16 + rlo;

    const _Float16* xrow = x16 + (size_t)rowA * FIN;
    const _Float16* WTh  = WT + (size_t)h * FOUT * FIN;

    constexpr int NOT = FOUT / 16;
    v8f c[NOT];
#pragma unroll
    for (int ot = 0; ot < NOT; ++ot) c[ot] = (v8f){0,0,0,0,0,0,0,0};

    for (int kt = 0; kt < FIN / 32; ++kt) {
        const int kb = kt * 32 + hi * 16;
        v8h alo = *(const v8h*)(xrow + kb);        // 16B
        v8h ahi = *(const v8h*)(xrow + kb + 8);    // 16B
        v16h a  = cat8(alo, ahi);
        const int kB = kt * 32 + hi * 16;
#pragma unroll
        for (int ot = 0; ot < NOT; ++ot) {
            const int col = ot * 16 + rlo;
            v16h b = *(const v16h*)(WTh + (size_t)col * FIN + kB);  // 32B
            c[ot] = wmma_f16(a, b, c[ot]);
        }
    }
#pragma unroll
    for (int ot = 0; ot < NOT; ++ot) {
        const int col = ot * 16 + rlo;
#pragma unroll
        for (int r = 0; r < 8; ++r) {
            const int row = tile * 16 + r + hi * 8;
            const float v = c[ot][r];
            Wh [((size_t)h * n + row) * FOUT + col] = v;
            WhT[((size_t)h * FOUT + col) * n + row] = (_Float16)v;
        }
    }
}

// ---------------------------------------------------------------------------
// Kernel 2: e_src[h,n] = Wh[h,n,:AD].a_src[h];  e_dst likewise.
// ---------------------------------------------------------------------------
template<int FOUT, int AD>
__global__ void attn_prep(
    const float* __restrict__ Wh, const float* __restrict__ asrc,
    const float* __restrict__ adst, float* __restrict__ es,
    float* __restrict__ ed, int n)
{
    const int idx = blockIdx.x * blockDim.x + threadIdx.x;
    if (idx >= NHEADS * n) return;
    const int h = idx / n, node = idx % n;
    const float* row = Wh + ((size_t)h * n + node) * FOUT;
    const float* as = asrc + h * AD;
    const float* ad = adst + h * AD;
    float s = 0.f, d = 0.f;
#pragma unroll 8
    for (int o = 0; o < AD; ++o) { float v = row[o]; s += v * as[o]; d += v * ad[o]; }
    es[idx] = s; ed[idx] = d;
}

// ---------------------------------------------------------------------------
// Kernel 3: per (h,row) masked online softmax stats over all m.
// One wave per row; coalesced adj reads; shfl_xor combine (wave32).
// ---------------------------------------------------------------------------
__global__ __launch_bounds__(256) void row_stats(
    const float* __restrict__ adj, const float* __restrict__ es,
    const float* __restrict__ ed, float* __restrict__ mx,
    float* __restrict__ inv, int n)
{
    const int wave = threadIdx.x >> 5;
    const int lane = threadIdx.x & 31;
    const int row  = blockIdx.x * 8 + wave;
    const int h    = blockIdx.y;
    const float* arow = adj + (size_t)row * n;
    const float* edh  = ed + (size_t)h * n;
    const float  esr  = es[(size_t)h * n + row];

    float m = -3.0e38f, s = 0.f;
    for (int j = lane; j < n; j += 32) {
        if (arow[j] > 0.f) {
            float e = esr + edh[j];
            e = (e > 0.f) ? e : 0.2f * e;
            float nm = fmaxf(m, e);
            s = s * __expf(m - nm) + __expf(e - nm);
            m = nm;
        }
    }
    for (int off = 16; off; off >>= 1) {
        float om = __shfl_xor(m, off, 32);
        float os = __shfl_xor(s, off, 32);
        float nm = fmaxf(m, om);
        s = s * __expf(m - nm) + os * __expf(om - nm);
        m = nm;
    }
    if (lane == 0) {
        mx [(size_t)h * n + row] = m;
        inv[(size_t)h * n + row] = 1.0f / s;
    }
}

// ---------------------------------------------------------------------------
// Kernel 4: out[n,:] = mean_h( softmax(masked e)[n,:] @ Wh[h] )  via WMMA.
// 4 waves per 16-row tile stride neighbor chunks of 32; deterministic
// fixed-order LDS reduction across waves; head-mean (+ReLU, +f16 copy) fused.
// global_prefetch of each wave's next adjacency chunk hides L2 latency.
// ---------------------------------------------------------------------------
template<int FOUT, bool RELU>
__global__ __launch_bounds__(128) void agg(
    const float*    __restrict__ adj,
    const _Float16* __restrict__ WhT,   // [H, FOUT, n]
    const float* __restrict__ es, const float* __restrict__ ed,
    const float* __restrict__ mx, const float* __restrict__ inv,
    float*    __restrict__ out,         // [n, FOUT]
    _Float16* __restrict__ out16,       // [n, FOUT] f16 (used when RELU)
    int n)
{
    constexpr int NOT = FOUT / 16;
    constexpr int EPT = (16 * FOUT) / 128;   // output elements per thread
    __shared__ float part[4][16][FOUT];

    const int tid  = threadIdx.x;
    const int w    = tid >> 5;
    const int lane = tid & 31;
    const int rlo  = lane & 15;
    const int hi   = lane >> 4;
    const int tile = blockIdx.x;
    const int rowA = tile * 16 + rlo;
    const float* arow = adj + (size_t)rowA * n;

    float acc[EPT];
#pragma unroll
    for (int e = 0; e < EPT; ++e) acc[e] = 0.f;

    for (int h = 0; h < NHEADS; ++h) {
        const float esr = es [(size_t)h * n + rowA];
        const float mxr = mx [(size_t)h * n + rowA];
        const float ivr = inv[(size_t)h * n + rowA];
        const float* edh = ed + (size_t)h * n;
        const _Float16* Bh = WhT + (size_t)h * FOUT * n;

        v8f c[NOT];
#pragma unroll
        for (int ot = 0; ot < NOT; ++ot) c[ot] = (v8f){0,0,0,0,0,0,0,0};

        for (int mc = w; mc < n / 32; mc += 4) {
            const int mb = mc * 32;
            // prefetch this wave's next adjacency chunk (one iteration ahead)
            if (mb + 128 < n) __builtin_prefetch(arow + mb + 128, 0, 0);
            const int kb = mb + hi * 16;
            v16h a;
#pragma unroll
            for (int j = 0; j < 8; ++j) {
                float e0 = esr + edh[kb + j];
                e0 = (e0 > 0.f) ? e0 : 0.2f * e0;
                float p0 = (arow[kb + j] > 0.f) ? __expf(e0 - mxr) * ivr : 0.f;
                a[j] = (_Float16)p0;
                float e1 = esr + edh[kb + 8 + j];
                e1 = (e1 > 0.f) ? e1 : 0.2f * e1;
                float p1 = (arow[kb + 8 + j] > 0.f) ? __expf(e1 - mxr) * ivr : 0.f;
                a[8 + j] = (_Float16)p1;
            }
            const int kB = mb + hi * 16;
#pragma unroll
            for (int ot = 0; ot < NOT; ++ot) {
                const int col = ot * 16 + rlo;
                v16h b = *(const v16h*)(Bh + (size_t)col * n + kB); // 32B aligned
                c[ot] = wmma_f16(a, b, c[ot]);
            }
        }
        // wave partials -> LDS (C layout: row r+8*hi, col 16*ot+rlo)
#pragma unroll
        for (int ot = 0; ot < NOT; ++ot)
#pragma unroll
            for (int r = 0; r < 8; ++r)
                part[w][r + hi * 8][ot * 16 + rlo] = c[ot][r];
        __syncthreads();
#pragma unroll
        for (int e = 0; e < EPT; ++e) {
            const int elem = tid * EPT + e;
            const int rr = elem / FOUT, cc = elem % FOUT;
            acc[e] += (part[0][rr][cc] + part[1][rr][cc]) +
                      (part[2][rr][cc] + part[3][rr][cc]);
        }
        __syncthreads();
    }
#pragma unroll
    for (int e = 0; e < EPT; ++e) {
        const int elem = tid * EPT + e;
        const int rr = elem / FOUT, cc = elem % FOUT;
        float v = acc[e] * (1.0f / 3.0f);
        if (RELU) v = fmaxf(v, 0.f);
        const size_t oidx = ((size_t)tile * 16 + rr) * FOUT + cc;
        out[oidx] = v;
        if (RELU) out16[oidx] = (_Float16)v;   // f16 input for layer-2 GEMM
    }
}

// ---------------------------------------------------------------------------
// Kernel 5: final log_softmax over 40 classes (input padded to 48 cols).
// ---------------------------------------------------------------------------
__global__ void logsoftmax40(const float* __restrict__ xin, float* __restrict__ y, int n)
{
    const int node = blockIdx.x * blockDim.x + threadIdx.x;
    if (node >= n) return;
    const float* r = xin + (size_t)node * 48;
    float m = -3.0e38f;
#pragma unroll
    for (int c = 0; c < 40; ++c) m = fmaxf(m, r[c]);
    float s = 0.f;
#pragma unroll
    for (int c = 0; c < 40; ++c) s += __expf(r[c] - m);
    const float ls = __logf(s) + m;
#pragma unroll
    for (int c = 0; c < 40; ++c) y[(size_t)node * 40 + c] = r[c] - ls;
}

// ---------------------------------------------------------------------------
extern "C" void kernel_launch(void* const* d_in, const int* in_sizes, int n_in,
                              void* d_out, int out_size, void* d_ws, size_t ws_size,
                              hipStream_t stream)
{
    const int n = NNODES;
    const float* feature = (const float*)d_in[0];  // [6144,512]
    const float* adj     = (const float*)d_in[1];  // [6144,6144]
    const float* W1      = (const float*)d_in[2];  // [3,512,64]
    const float* a1_src  = (const float*)d_in[3];  // [3,64]
    const float* a1_dst  = (const float*)d_in[4];
    const float* W2      = (const float*)d_in[5];  // [3,64,40]
    const float* a2_src  = (const float*)d_in[6];  // [3,40]
    const float* a2_dst  = (const float*)d_in[7];

    // ---- workspace carve-up (~23 MB, all chunks multiples of 256B) ----
    char* ws = (char*)d_ws;
    size_t o = 0;
    _Float16* x16   = (_Float16*)(ws + o); o += (size_t)n * 512 * 2;        // feature f16
    _Float16* WT1   = (_Float16*)(ws + o); o += (size_t)3 * 64 * 512 * 2;   // W1^T f16
    _Float16* WT2   = (_Float16*)(ws + o); o += (size_t)3 * 48 * 64 * 2;    // W2^T f16 (padded)
    float*    Wh1   = (float*)   (ws + o); o += (size_t)3 * n * 64 * 4;
    _Float16* WhT1  = (_Float16*)(ws + o); o += (size_t)3 * 64 * n * 2;
    float*    es1   = (float*)   (ws + o); o += (size_t)3 * n * 4;
    float*    ed1   = (float*)   (ws + o); o += (size_t)3 * n * 4;
    float*    mx1   = (float*)   (ws + o); o += (size_t)3 * n * 4;
    float*    inv1  = (float*)   (ws + o); o += (size_t)3 * n * 4;
    float*    x2    = (float*)   (ws + o); o += (size_t)n * 64 * 4;
    _Float16* x2_16 = (_Float16*)(ws + o); o += (size_t)n * 64 * 2;
    float*    Wh2   = (float*)   (ws + o); o += (size_t)3 * n * 48 * 4;
    _Float16* WhT2  = (_Float16*)(ws + o); o += (size_t)3 * 48 * n * 2;
    float*    es2   = (float*)   (ws + o); o += (size_t)3 * n * 4;
    float*    ed2   = (float*)   (ws + o); o += (size_t)3 * n * 4;
    float*    mx2   = (float*)   (ws + o); o += (size_t)3 * n * 4;
    float*    inv2  = (float*)   (ws + o); o += (size_t)3 * n * 4;
    float*    out2  = (float*)   (ws + o); o += (size_t)n * 48 * 4;
    (void)ws_size; (void)in_sizes; (void)n_in; (void)out_size;

    const int ntiles = n / 16;  // 384

    // ---------------- prep (f16 conversions, weight transposes) -------------
    cvt_f16<<<(n * 512 + 255) / 256, 256, 0, stream>>>(feature, x16, n * 512);
    wt_prep<512, 64, 64><<<(NHEADS * 64 * 512 + 255) / 256, 256, 0, stream>>>(W1, WT1);
    wt_prep<64, 48, 40><<<(NHEADS * 48 * 64 + 255) / 256, 256, 0, stream>>>(W2, WT2);

    // ---------------- layer 1 ----------------
    wh_gemm<512, 64><<<dim3(ntiles, NHEADS), 32, 0, stream>>>(x16, WT1, Wh1, WhT1, n);
    attn_prep<64, 64><<<(NHEADS * n + 255) / 256, 256, 0, stream>>>(Wh1, a1_src, a1_dst, es1, ed1, n);
    row_stats<<<dim3(n / 8, NHEADS), 256, 0, stream>>>(adj, es1, ed1, mx1, inv1, n);
    agg<64, true><<<ntiles, 128, 0, stream>>>(adj, WhT1, es1, ed1, mx1, inv1, x2, x2_16, n);

    // ---------------- layer 2 ----------------
    wh_gemm<64, 48><<<dim3(ntiles, NHEADS), 32, 0, stream>>>(x2_16, WT2, Wh2, WhT2, n);
    attn_prep<48, 40><<<(NHEADS * n + 255) / 256, 256, 0, stream>>>(Wh2, a2_src, a2_dst, es2, ed2, n);
    row_stats<<<dim3(n / 8, NHEADS), 256, 0, stream>>>(adj, es2, ed2, mx2, inv2, n);
    agg<48, false><<<ntiles, 128, 0, stream>>>(adj, WhT2, es2, ed2, mx2, inv2, out2, (_Float16*)nullptr, n);

    // ---------------- final log_softmax ----------------
    logsoftmax40<<<(n + 255) / 256, 256, 0, stream>>>(out2, (float*)d_out, n);
}
